// AdaptiveAttentionHead_23837068493161
// MI455X (gfx1250) — matrix operand order repaired and checked by reading, not compiled, not run on hardware
//
#include <hip/hip_runtime.h>

// ---------------------------------------------------------------------------
// Problem constants (from reference): x:(4,4096,2048) f32, W*:(128,2048) f32
// ---------------------------------------------------------------------------
#define B_DIM 4
#define T_DIM 4096
#define C_DIM 2048
#define D_DIM 128
#define BT    (B_DIM * T_DIM)

typedef _Float16 h8  __attribute__((ext_vector_type(8)));
typedef _Float16 h16 __attribute__((ext_vector_type(16)));
typedef float    f8  __attribute__((ext_vector_type(8)));

#ifndef __has_builtin
#define __has_builtin(x) 0
#endif

static __device__ __forceinline__ h16 cat_h8(h8 lo, h8 hi) {
  return __builtin_shufflevector(lo, hi, 0,1,2,3,4,5,6,7,8,9,10,11,12,13,14,15);
}

// D(f32 16x16) = A(f16 16x32) * B(f16 32x16) + C
static __device__ __forceinline__ f8 wmma_f16(h16 a, h16 b, f8 c) {
  return __builtin_amdgcn_wmma_f32_16x16x32_f16(false, a, false, b, (short)0, c,
                                                false, false);
}

// 16-byte global -> LDS copy. Prefer the CDNA5 async-DMA path
// (GLOBAL_LOAD_ASYNC_TO_LDS_B128, tracked by ASYNCcnt); fall back to a plain
// VGPR round trip if the builtin is unavailable.
// Probe-derived signature: (v4i addrspace(1)*, v4i addrspace(3)*, i32, i32).
typedef int i4v __attribute__((vector_size(16)));
typedef __attribute__((address_space(1))) i4v* gptr_i4v;
typedef __attribute__((address_space(3))) i4v* lptr_i4v;

static __device__ __forceinline__ void copy16_g2lds(const void* gsrc,
                                                    void* lds_dst) {
#if __has_builtin(__builtin_amdgcn_global_load_async_to_lds_b128)
  __builtin_amdgcn_global_load_async_to_lds_b128(
      (gptr_i4v)gsrc, (lptr_i4v)lds_dst, 0, 0);
#else
  *(int4*)lds_dst = *(const int4*)gsrc;
#endif
}

static __device__ __forceinline__ void wait_async_copies() {
#if __has_builtin(__builtin_amdgcn_s_wait_asynccnt)
  __builtin_amdgcn_s_wait_asynccnt(0);
#elif defined(__AMDGCN__)
  asm volatile("s_wait_asynccnt 0" ::: "memory");
#endif
}

// ---------------------------------------------------------------------------
// Kernel 1: fp32 -> fp16 conversion (weights; 1.5 MB total -> L2 resident)
// ---------------------------------------------------------------------------
__global__ void cvt_f32_to_f16(const float* __restrict__ in,
                               _Float16* __restrict__ out, int n) {
  int i = blockIdx.x * blockDim.x + threadIdx.x;
  if (i < n) out[i] = (_Float16)in[i];
}

// ---------------------------------------------------------------------------
// Kernel 2: QKV projection.
// Block = 64 tokens x 128 dims (8 waves). Wave w: token subtile (w&3),
// d half (w>>2): 16 tokens x 64 dims x {Q,K,V} = 12 accum tiles,
// 12 WMMAs per 32-wide K step, ONE barrier per step (double-buffered x tile,
// staged f32 via async global->LDS, converted to f16 during A-frag assembly).
//
// WMMA operand layouts (ISA 7.12.2), lane m/n = lane&15, h = lane>>4:
//   A (16x32 f16): a[0..7] = K h*8..h*8+7 ; a[8..15] = K 16+h*8..16+h*8+7
//   B (32x16 f16): col n, b[0..7] = K h*8..h*8+7 ; b[8..15] = K 16+h*8..16+h*8+7
//   C (f32)      : vgpr r -> row (r + 8h), col n
// ---------------------------------------------------------------------------
__global__ __launch_bounds__(256) void qkv_gemm(
    const float* __restrict__ x,
    const _Float16* __restrict__ Wqh, const _Float16* __restrict__ Wkh,
    const _Float16* __restrict__ Wvh,
    const float* __restrict__ bq, const float* __restrict__ bk,
    const float* __restrict__ bv,
    _Float16* __restrict__ Qh, _Float16* __restrict__ Kh,
    _Float16* __restrict__ Vh)
{
  __shared__ __attribute__((aligned(16))) float xs[2][64][32];  // 16 KB

  const int tid  = threadIdx.x;
  const int lane = tid & 31;
  const int wv   = tid >> 5;
  const int n    = lane & 15;
  const int hh   = (lane >> 4) & 1;
  const int tt   = wv & 3;   // token subtile within block
  const int dh   = wv >> 2;  // d half: dims [64*dh, 64*dh+64)
  const int tok0 = blockIdx.x * 64;

  const _Float16* Ws[3] = {Wqh, Wkh, Wvh};

  const f8 fzero = {};
  f8 acc[3][4];
#pragma unroll
  for (int mt = 0; mt < 3; ++mt)
#pragma unroll
    for (int dt = 0; dt < 4; ++dt) acc[mt][dt] = fzero;

  // cooperative stage of one 64x32 f32 x-tile: 8 floats (2x16B) per thread
  const int se = tid * 8;
  const int sr = se >> 5, sc = se & 31;

  { // prologue: stage k0 = 0 into buffer 0
    const float* g = x + (size_t)(tok0 + sr) * C_DIM + sc;
    copy16_g2lds(g, &xs[0][sr][sc]);
    copy16_g2lds(g + 4, &xs[0][sr][sc + 4]);
  }

  const int nsteps = C_DIM / 32;
  for (int it = 0; it < nsteps; ++it) {
    const int k0  = it * 32;
    const int buf = it & 1;

    wait_async_copies();
    __syncthreads();  // buf data visible; previous reads of buf^1 finished

    if (it + 1 < nsteps) {  // stage next tile into the other buffer
      const float* g = x + (size_t)(tok0 + sr) * C_DIM + (k0 + 32) + sc;
      copy16_g2lds(g, &xs[buf ^ 1][sr][sc]);
      copy16_g2lds(g + 4, &xs[buf ^ 1][sr][sc + 4]);
    }

    // A fragment: read f32 from LDS, convert to f16 in registers
    const float* xr = &xs[buf][tt * 16 + n][0];
    float4 p0 = *(const float4*)(xr + hh * 8);
    float4 p1 = *(const float4*)(xr + hh * 8 + 4);
    float4 p2 = *(const float4*)(xr + 16 + hh * 8);
    float4 p3 = *(const float4*)(xr + 16 + hh * 8 + 4);
    h16 a;
    a[0]  = (_Float16)p0.x; a[1]  = (_Float16)p0.y;
    a[2]  = (_Float16)p0.z; a[3]  = (_Float16)p0.w;
    a[4]  = (_Float16)p1.x; a[5]  = (_Float16)p1.y;
    a[6]  = (_Float16)p1.z; a[7]  = (_Float16)p1.w;
    a[8]  = (_Float16)p2.x; a[9]  = (_Float16)p2.y;
    a[10] = (_Float16)p2.z; a[11] = (_Float16)p2.w;
    a[12] = (_Float16)p3.x; a[13] = (_Float16)p3.y;
    a[14] = (_Float16)p3.z; a[15] = (_Float16)p3.w;

#pragma unroll
    for (int mt = 0; mt < 3; ++mt)
#pragma unroll
      for (int dt = 0; dt < 4; ++dt) {
        const int d = dh * 64 + dt * 16 + n;
        h16 bfrag = cat_h8(*(const h8*)(Ws[mt] + (size_t)d * C_DIM + k0 + hh * 8),
                           *(const h8*)(Ws[mt] + (size_t)d * C_DIM + k0 + 16 + hh * 8));
        acc[mt][dt] = wmma_f16(a, bfrag, acc[mt][dt]);
      }
  }

  _Float16* QKV[3] = {Qh, Kh, Vh};
  const float* Bs[3] = {bq, bk, bv};
#pragma unroll
  for (int mt = 0; mt < 3; ++mt)
#pragma unroll
    for (int dt = 0; dt < 4; ++dt) {
      const int d = dh * 64 + dt * 16 + n;
      const float bias = Bs[mt][d];
#pragma unroll
      for (int r = 0; r < 8; ++r) {
        int row = tt * 16 + r + 8 * hh;
        QKV[mt][(size_t)(tok0 + row) * D_DIM + d] =
            (_Float16)(acc[mt][dt][r] + bias);
      }
    }
}

// ---------------------------------------------------------------------------
// Kernel 3: RoPE in place on Q and K (pair (d, d+64), angle = t * base^{-d/64})
// ---------------------------------------------------------------------------
__global__ void rope_inplace(_Float16* __restrict__ Qh,
                             _Float16* __restrict__ Kh) {
  int idx = blockIdx.x * blockDim.x + threadIdx.x;
  if (idx >= BT * 64) return;
  int dh  = idx & 63;
  int tok = idx >> 6;
  int t   = tok & (T_DIM - 1);
  float invf = __expf(-logf(10000.0f) * (float)dh * (1.0f / 64.0f));
  float ang  = (float)t * invf;
  float cs = cosf(ang), sn = sinf(ang);
  {
    _Float16* q = Qh + (size_t)tok * D_DIM;
    float a = (float)q[dh], b2 = (float)q[dh + 64];
    q[dh]      = (_Float16)(a * cs - b2 * sn);
    q[dh + 64] = (_Float16)(b2 * cs + a * sn);
  }
  {
    _Float16* k = Kh + (size_t)tok * D_DIM;
    float a = (float)k[dh], b2 = (float)k[dh + 64];
    k[dh]      = (_Float16)(a * cs - b2 * sn);
    k[dh + 64] = (_Float16)(b2 * cs + a * sn);
  }
}

// ---------------------------------------------------------------------------
// Kernel 4: causal flash attention. grid = (T/128, B), block = 256 (8 waves).
// Wave w owns query rows [qblk*128 + 16w, +16). Per 32-key block:
//   S(16x32)  = Q(16x128) . K^T   -> 2 subtiles x 4 WMMAs
//   online softmax (shfl_xor row reductions across the 16-lane halves)
//   O(16x128) += P(16x32 f16) . V -> 8 WMMAs
// K tile staged via async global->LDS; V transposed to d-major through VGPRs;
// P re-staged through per-wave LDS to convert C-layout -> A-layout.
// ---------------------------------------------------------------------------
__global__ __launch_bounds__(256) void flash_attn(
    const _Float16* __restrict__ Qh, const _Float16* __restrict__ Kh,
    const _Float16* __restrict__ Vh, float* __restrict__ out)
{
  __shared__ __attribute__((aligned(64))) _Float16 Ks[32][128];  // key-major
  __shared__ __attribute__((aligned(64))) _Float16 Vs[128][32];  // d-major
  __shared__ __attribute__((aligned(64))) _Float16 Ps[8][16][32];

  const int b    = blockIdx.y;
  const int qblk = blockIdx.x;
  const int tid  = threadIdx.x;
  const int lane = tid & 31;
  const int wv   = tid >> 5;
  const int n    = lane & 15;
  const int hh   = (lane >> 4) & 1;
  const int q0   = qblk * 128 + wv * 16;

  const size_t bbase = (size_t)b * T_DIM * D_DIM;
  const _Float16* Qb = Qh + bbase;
  const _Float16* Kb = Kh + bbase;
  const _Float16* Vb = Vh + bbase;

  // Q fragments (A layout), K-dim 128 = 4 x 32
  h16 aq[4];
#pragma unroll
  for (int kk = 0; kk < 4; ++kk) {
    const _Float16* p = Qb + (size_t)(q0 + n) * D_DIM + kk * 32 + hh * 8;
    aq[kk] = cat_h8(*(const h8*)p, *(const h8*)(p + 16));
  }

  const f8 fzero = {};
  f8 o[8];
  float m8[8], l8[8];
#pragma unroll
  for (int dt = 0; dt < 8; ++dt) o[dt] = fzero;
#pragma unroll
  for (int r = 0; r < 8; ++r) { m8[r] = -INFINITY; l8[r] = 0.0f; }

  const float scale = 0.08838834764831845f;  // 1/sqrt(128)
  const int jmax = qblk * 128 + 128;

  // staging split: thread owns 16 f16 of K (2x16B async) and 16 f16 of V
  const int e  = tid * 16;
  const int kr = e >> 7;
  const int kc = e & 127;

  for (int j0 = 0; j0 < jmax; j0 += 32) {
    __syncthreads();  // previous iteration's LDS readers are done
    {
      const _Float16* kg = Kb + (size_t)(j0 + kr) * D_DIM + kc;
      copy16_g2lds(kg, &Ks[kr][kc]);
      copy16_g2lds(kg + 8, &Ks[kr][kc + 8]);

      h16 vvec = *(const h16*)(Vb + (size_t)(j0 + kr) * D_DIM + kc);
#pragma unroll
      for (int i = 0; i < 16; ++i) Vs[kc + i][kr] = vvec[i];
    }
    wait_async_copies();
    __syncthreads();

    if (j0 <= q0 + 15) {  // wave-uniform: EXEC stays all-ones for WMMA
      f8 s[2];
#pragma unroll
      for (int sub = 0; sub < 2; ++sub) {
        f8 acc = fzero;
#pragma unroll
        for (int kk = 0; kk < 4; ++kk) {
          h16 fbk = cat_h8(*(const h8*)(&Ks[sub * 16 + n][kk * 32 + hh * 8]),
                           *(const h8*)(&Ks[sub * 16 + n][kk * 32 + 16 + hh * 8]));
          acc = wmma_f16(aq[kk], fbk, acc);
        }
        int key = j0 + sub * 16 + n;
#pragma unroll
        for (int r = 0; r < 8; ++r) {
          float v = acc[r] * scale;
          int qrow = q0 + r + 8 * hh;
          s[sub][r] = (key > qrow) ? -INFINITY : v;
        }
      }

      // online softmax per row (rows live across 16 lanes of a half-wave)
#pragma unroll
      for (int r = 0; r < 8; ++r) {
        float v = fmaxf(s[0][r], s[1][r]);
#pragma unroll
        for (int off = 8; off >= 1; off >>= 1)
          v = fmaxf(v, __shfl_xor(v, off, 32));
        float mn    = fmaxf(m8[r], v);
        float alpha = __expf(m8[r] - mn);
        float p0 = __expf(s[0][r] - mn);
        float p1 = __expf(s[1][r] - mn);
        s[0][r] = p0; s[1][r] = p1;
        float rs = p0 + p1;
#pragma unroll
        for (int off = 8; off >= 1; off >>= 1)
          rs += __shfl_xor(rs, off, 32);
        l8[r] = l8[r] * alpha + rs;
        m8[r] = mn;
#pragma unroll
        for (int dt = 0; dt < 8; ++dt) o[dt][r] *= alpha;
      }

      // C-layout -> A-layout via per-wave LDS staging
#pragma unroll
      for (int sub = 0; sub < 2; ++sub)
#pragma unroll
        for (int r = 0; r < 8; ++r)
          Ps[wv][r + 8 * hh][sub * 16 + n] = (_Float16)s[sub][r];

      asm volatile("s_wait_dscnt 0" ::: "memory");  // wave-local LDS RAW

      h16 ap = cat_h8(*(const h8*)(&Ps[wv][n][hh * 8]),
                      *(const h8*)(&Ps[wv][n][16 + hh * 8]));

#pragma unroll
      for (int dt = 0; dt < 8; ++dt) {
        h16 fbv = cat_h8(*(const h8*)(&Vs[dt * 16 + n][hh * 8]),
                         *(const h8*)(&Vs[dt * 16 + n][16 + hh * 8]));
        o[dt] = wmma_f16(ap, fbv, o[dt]);
      }
    }
  }

#pragma unroll
  for (int r = 0; r < 8; ++r) {
    float inv = 1.0f / l8[r];
    int row = q0 + r + 8 * hh;
#pragma unroll
    for (int dt = 0; dt < 8; ++dt) {
      out[(size_t)(b * T_DIM + row) * D_DIM + dt * 16 + n] = o[dt][r] * inv;
    }
  }
}

// ---------------------------------------------------------------------------
// Launch
// ---------------------------------------------------------------------------
extern "C" void kernel_launch(void* const* d_in, const int* in_sizes, int n_in,
                              void* d_out, int out_size, void* d_ws,
                              size_t ws_size, hipStream_t stream) {
  (void)in_sizes; (void)n_in; (void)out_size; (void)ws_size;

  const float* x  = (const float*)d_in[0];
  const float* Wq = (const float*)d_in[1];
  const float* Wk = (const float*)d_in[2];
  const float* Wv = (const float*)d_in[3];
  const float* bq = (const float*)d_in[4];
  const float* bk = (const float*)d_in[5];
  const float* bv = (const float*)d_in[6];
  float* out = (float*)d_out;

  // Workspace layout (f16): Wq|Wk|Wv (0.5 MB each) then Q|K|V (4 MB each)
  char* ws = (char*)d_ws;
  const size_t WN = (size_t)D_DIM * C_DIM;  // 262144
  const size_t QN = (size_t)BT * D_DIM;     // 2097152
  _Float16* Wqh = (_Float16*)ws;
  _Float16* Wkh = Wqh + WN;
  _Float16* Wvh = Wkh + WN;
  _Float16* Qh  = Wvh + WN;
  _Float16* Kh  = Qh + QN;
  _Float16* Vh  = Kh + QN;

  {
    int nw = (int)WN;
    int blocks = (nw + 255) / 256;
    cvt_f32_to_f16<<<blocks, 256, 0, stream>>>(Wq, Wqh, nw);
    cvt_f32_to_f16<<<blocks, 256, 0, stream>>>(Wk, Wkh, nw);
    cvt_f32_to_f16<<<blocks, 256, 0, stream>>>(Wv, Wvh, nw);
  }

  qkv_gemm<<<BT / 64, 256, 0, stream>>>(x, Wqh, Wkh, Wvh, bq, bk, bv,
                                        Qh, Kh, Vh);

  {
    int nr = BT * 64;
    rope_inplace<<<(nr + 255) / 256, 256, 0, stream>>>(Qh, Kh);
  }

  flash_attn<<<dim3(T_DIM / 128, B_DIM), 256, 0, stream>>>(Qh, Kh, Vh, out);
}
